// DifferentiableFluidSimulator_65953517797804
// MI455X (gfx1250) — matrix operand order repaired and checked by reading, not compiled, not run on hardware
//
#include <hip/hip_runtime.h>
#include <hip/hip_bf16.h>
#include <math.h>

// ---------------- constants ----------------
constexpr int   G    = 96;
constexpr int   NV   = G * G * G;           // 884736
constexpr float DT   = 0.01f;
constexpr float VISC = 0.001f;

// ---------------- vector types ----------------
typedef __attribute__((ext_vector_type(16))) _Float16 v16h;
typedef __attribute__((ext_vector_type(8)))  float    v8f;
typedef __attribute__((ext_vector_type(8)))  unsigned int v8u;

union VU {
    v8u  u;
    v16h h;
    unsigned int ui[8];
    _Float16     hh[16];
};
union HP { _Float16 h[2]; unsigned int u; };

// ---------------- gradient helpers (np.gradient semantics) ----------------
__device__ __forceinline__ float fld(const float* f, int x, int y, int z) {
    return f[(x * G + y) * G + z];
}

__device__ __forceinline__ float grad1(const float* f, int x, int y, int z, int axis) {
    int i = (axis == 0) ? x : (axis == 1) ? y : z;
    int ip, im; float s;
    if (i == 0)            { ip = 1;     im = 0;     s = 1.0f; }
    else if (i == G - 1)   { ip = G - 1; im = G - 2; s = 1.0f; }
    else                   { ip = i + 1; im = i - 1; s = 0.5f; }
    float fp, fm;
    if (axis == 0)      { fp = fld(f, ip, y, z); fm = fld(f, im, y, z); }
    else if (axis == 1) { fp = fld(f, x, ip, z); fm = fld(f, x, im, z); }
    else                { fp = fld(f, x, y, ip); fm = fld(f, x, y, im); }
    return (fp - fm) * s;
}

__device__ __forceinline__ float grad1_at(const float* f, int x, int y, int z, int axis, int i) {
    if (axis == 0) x = i; else if (axis == 1) y = i; else z = i;
    return grad1(f, x, y, z, axis);
}

// gradient-of-gradient (reference's "laplacian" per axis)
__device__ __forceinline__ float grad2(const float* f, int x, int y, int z, int axis) {
    int i = (axis == 0) ? x : (axis == 1) ? y : z;
    if (i == 0)     return grad1_at(f, x, y, z, axis, 1)     - grad1_at(f, x, y, z, axis, 0);
    if (i == G - 1) return grad1_at(f, x, y, z, axis, G - 1) - grad1_at(f, x, y, z, axis, G - 2);
    return 0.5f * (grad1_at(f, x, y, z, axis, i + 1) - grad1_at(f, x, y, z, axis, i - 1));
}

// ---------------- K0: pack weights into B-matrix fragment layout ----------------
// 58 fragments, each: 32 lanes x 8 dwords (16 f16 halves) = exact VGPR image of a
// 32x16 f16 B operand.  B layout: lane holds column n = lane%16;
// K = kTile*32 + (lane>=16 ? 8 : 0) + (v>=4 ? 16 : 0) + 2*(v&3) + h   (v = dword 0..7, h = half 0..1)
// frag table: [0,8)=w1(nt=f)  [8,40)=w2(nt=(f-8)/4,kt=(f-8)%4)
//             [40,56)=w3      [56,58)=w4(kt=f-56)
__global__ void pack_weights_kernel(const float* __restrict__ w1,
                                    const float* __restrict__ w2,
                                    const float* __restrict__ w3,
                                    const float* __restrict__ w4,
                                    unsigned int* __restrict__ wpack) {
    int t = blockIdx.x * blockDim.x + threadIdx.x;
    if (t >= 58 * 256) return;
    int f    = t >> 8;
    int r    = t & 255;
    int lane = r >> 3;
    int v    = r & 7;

    const float* W; int K, Nn, nt, kt;
    if (f < 8)       { W = w1; K = 4;   Nn = 128; nt = f;             kt = 0;          }
    else if (f < 40) { W = w2; K = 128; Nn = 128; nt = (f - 8) >> 2;  kt = (f - 8) & 3; }
    else if (f < 56) { W = w3; K = 128; Nn = 64;  nt = (f - 40) >> 2; kt = (f - 40) & 3; }
    else             { W = w4; K = 64;  Nn = 3;   nt = 0;             kt = f - 56;     }

    int n     = nt * 16 + (lane & 15);
    int kbase = kt * 32 + ((lane & 16) ? 8 : 0) + ((v >= 4) ? 16 : 0) + 2 * (v & 3);

    HP p; p.h[0] = (_Float16)0.0f; p.h[1] = (_Float16)0.0f;
    if (n < Nn) {
        if (kbase     < K) p.h[0] = (_Float16)W[(size_t)kbase       * Nn + n];
        if (kbase + 1 < K) p.h[1] = (_Float16)W[(size_t)(kbase + 1) * Nn + n];
    }
    wpack[((size_t)f * 32 + lane) * 8 + v] = p.u;
}

// ---------------- K1: advect velocity + density, apply sources ----------------
__global__ void advect_kernel(const float* __restrict__ rho_in,
                              const float* __restrict__ vel_in,
                              const float* __restrict__ src,
                              float* __restrict__ rho_out,
                              float* __restrict__ vel_ws) {
    int idx = blockIdx.x * blockDim.x + threadIdx.x;
    if (idx >= NV) return;
    int z = idx % G, y = (idx / G) % G, x = idx / (G * G);

    float ds = grad1(rho_in, x, y, z, 0) + grad1(rho_in, x, y, z, 1) + grad1(rho_in, x, y, z, 2);
    float rr = rho_in[idx];
    rho_out[idx] = rr - rr * ds * DT + src[0 * NV + idx] * DT;

#pragma unroll
    for (int c = 0; c < 3; ++c) {
        const float* f = vel_in + (size_t)c * NV;
        float g = grad1(f, x, y, z, 0) + grad1(f, x, y, z, 1) + grad1(f, x, y, z, 2);
        float vv = f[idx];
        vel_ws[(size_t)c * NV + idx] = vv - vv * g + src[(size_t)(1 + c) * NV + idx] * DT;
    }
}

// ---------------- K2: per-voxel MLP via WMMA ----------------
__device__ __forceinline__ v8f wmma32(v16h a, v16h b, v8f c) {
    return __builtin_amdgcn_wmma_f32_16x16x32_f16(false, a, false, b, (short)0, c, false, false);
}

__device__ __forceinline__ v16h load_B(const unsigned int* __restrict__ wpack, int frag, int lane) {
    VU r;
    r.u = *(const v8u*)(wpack + ((size_t)frag * 32 + lane) * 8);
    return r.h;
}

// A-matrix f16 16x32 layout (ISA 7.12.2): lane m=lane%16;
// K = kt*32 + (lane>=16?8:0) + (v>=4?16:0) + 2*(v&3) + half
// VGPRs 0..3 hold halves K=koff..koff+7 (16 contiguous bytes) and
// VGPRs 4..7 hold halves K=koff+16..koff+23 -> two ds_load_b128 per tile.
__device__ __forceinline__ v16h load_A_lds(const _Float16* __restrict__ Lb, int lane, int kt, int rowHalves) {
    int m    = lane & 15;
    int koff = kt * 32 + ((lane & 16) ? 8 : 0);
    const uint4* p0 = (const uint4*)(Lb + m * rowHalves + koff);
    const uint4* p1 = (const uint4*)(Lb + m * rowHalves + koff + 16);
    uint4 a = *p0;
    uint4 b = *p1;
    VU r;
    r.ui[0] = a.x; r.ui[1] = a.y; r.ui[2] = a.z; r.ui[3] = a.w;
    r.ui[4] = b.x; r.ui[5] = b.y; r.ui[6] = b.z; r.ui[7] = b.w;
    return r.h;
}

// D layout: vgpr j holds row M = j + (lane>=16 ? 8 : 0), column n = lane%16
__device__ __forceinline__ void store_D_lds(_Float16* __restrict__ Lb, int lane, int nt, v8f c, int rowHalves) {
    int n  = nt * 16 + (lane & 15);
    int mb = (lane & 16) ? 8 : 0;
#pragma unroll
    for (int j = 0; j < 8; ++j)
        Lb[(mb + j) * rowHalves + n] = (_Float16)c[j];
}

__device__ __forceinline__ v8f relu8(v8f c) {
#pragma unroll
    for (int j = 0; j < 8; ++j) c[j] = fmaxf(c[j], 0.0f);
    return c;
}

__device__ __forceinline__ v8f splat8(float b) {
    v8f c;
#pragma unroll
    for (int j = 0; j < 8; ++j) c[j] = b;
    return c;
}

__global__ __launch_bounds__(128) void mlp_kernel(float* __restrict__ vel,
                                                  const float* __restrict__ rho,
                                                  const unsigned int* __restrict__ wpack,
                                                  const float* __restrict__ b1,
                                                  const float* __restrict__ b2,
                                                  const float* __restrict__ b3,
                                                  const float* __restrict__ b4) {
    __shared__ __align__(16) _Float16 lbuf[4][16 * 128];
    __shared__ float tbuf[4][16][4];

    const int lane = threadIdx.x & 31;
    const int w    = threadIdx.x >> 5;
    const int tile = blockIdx.x * 4 + w;      // 55296 tiles exactly, all waves active
    const int base = tile * 16;
    _Float16* L = lbuf[w];
    const int cl = lane & 15;

    // ---- layer-1 A operand: feats [vx,vy,vz,rho], K padded 4->32 ----
    VU a1;
#pragma unroll
    for (int v = 0; v < 8; ++v) a1.ui[v] = 0u;
    if (lane < 16) {
        int vox = base + lane;
        a1.hh[0] = (_Float16)vel[0 * NV + vox];
        a1.hh[1] = (_Float16)vel[1 * NV + vox];
        a1.hh[2] = (_Float16)vel[2 * NV + vox];
        a1.hh[3] = (_Float16)rho[vox];
    }

    v8f acc[8];

    // ---- layer 1: [16,4] @ [4,128] ----
#pragma unroll
    for (int nt = 0; nt < 8; ++nt) {
        v8f c = splat8(b1[nt * 16 + cl]);
        c = wmma32(a1.h, load_B(wpack, nt, lane), c);
        acc[nt] = relu8(c);
    }
    __builtin_amdgcn_fence(__ATOMIC_ACQ_REL, "workgroup");
#pragma unroll
    for (int nt = 0; nt < 8; ++nt) store_D_lds(L, lane, nt, acc[nt], 128);
    __builtin_amdgcn_fence(__ATOMIC_ACQ_REL, "workgroup");

    v16h aT[4];
#pragma unroll
    for (int kt = 0; kt < 4; ++kt) aT[kt] = load_A_lds(L, lane, kt, 128);

    // ---- layer 2: [16,128] @ [128,128] ----
#pragma unroll
    for (int nt = 0; nt < 8; ++nt) {
        v8f c = splat8(b2[nt * 16 + cl]);
#pragma unroll
        for (int kt = 0; kt < 4; ++kt)
            c = wmma32(aT[kt], load_B(wpack, 8 + nt * 4 + kt, lane), c);
        acc[nt] = relu8(c);
    }
    __builtin_amdgcn_fence(__ATOMIC_ACQ_REL, "workgroup");
#pragma unroll
    for (int nt = 0; nt < 8; ++nt) store_D_lds(L, lane, nt, acc[nt], 128);
    __builtin_amdgcn_fence(__ATOMIC_ACQ_REL, "workgroup");
#pragma unroll
    for (int kt = 0; kt < 4; ++kt) aT[kt] = load_A_lds(L, lane, kt, 128);

    // ---- layer 3: [16,128] @ [128,64] ----
#pragma unroll
    for (int nt = 0; nt < 4; ++nt) {
        v8f c = splat8(b3[nt * 16 + cl]);
#pragma unroll
        for (int kt = 0; kt < 4; ++kt)
            c = wmma32(aT[kt], load_B(wpack, 40 + nt * 4 + kt, lane), c);
        acc[nt] = relu8(c);
    }
    __builtin_amdgcn_fence(__ATOMIC_ACQ_REL, "workgroup");
#pragma unroll
    for (int nt = 0; nt < 4; ++nt) store_D_lds(L, lane, nt, acc[nt], 64);
    __builtin_amdgcn_fence(__ATOMIC_ACQ_REL, "workgroup");
#pragma unroll
    for (int kt = 0; kt < 2; ++kt) aT[kt] = load_A_lds(L, lane, kt, 64);

    // ---- layer 4: [16,64] @ [64,3 padded 16] ----
    {
        v8f c = splat8((cl < 3) ? b4[cl] : 0.0f);
        c = wmma32(aT[0], load_B(wpack, 56, lane), c);
        c = wmma32(aT[1], load_B(wpack, 57, lane), c);

        if (cl < 3) {
            int mb = (lane & 16) ? 8 : 0;
#pragma unroll
            for (int j = 0; j < 8; ++j)
                tbuf[w][mb + j][cl] = tanhf(c[j]) * 0.1f;
        }
    }
    __builtin_amdgcn_fence(__ATOMIC_ACQ_REL, "workgroup");

    if (lane < 16) {
        int vox = base + lane;
#pragma unroll
        for (int c = 0; c < 3; ++c)
            vel[(size_t)c * NV + vox] += tbuf[w][lane][c] * DT;
    }
}

// ---------------- K3: divergence -> pressure ----------------
__global__ void pressure_kernel(const float* __restrict__ vel_ws,
                                const float* __restrict__ p_in,
                                float* __restrict__ p_out) {
    int idx = blockIdx.x * blockDim.x + threadIdx.x;
    if (idx >= NV) return;
    int z = idx % G, y = (idx / G) % G, x = idx / (G * G);
    float div = grad1(vel_ws + 0 * NV, x, y, z, 0)
              + grad1(vel_ws + 1 * NV, x, y, z, 1)
              + grad1(vel_ws + 2 * NV, x, y, z, 2);
    p_out[idx] = p_in[idx] + 0.1f * div;
}

// ---------------- K4: v -= grad(p)*DT (in place) ----------------
__global__ void subgrad_kernel(float* __restrict__ vel_ws,
                               const float* __restrict__ p) {
    int idx = blockIdx.x * blockDim.x + threadIdx.x;
    if (idx >= NV) return;
    int z = idx % G, y = (idx / G) % G, x = idx / (G * G);
#pragma unroll
    for (int c = 0; c < 3; ++c)
        vel_ws[(size_t)c * NV + idx] -= DT * grad1(p, x, y, z, c);
}

// ---------------- K5: diffuse (grad-of-grad laplacian) -> output ----------------
__global__ void diffuse_kernel(const float* __restrict__ vel_ws,
                               float* __restrict__ vel_out) {
    int idx = blockIdx.x * blockDim.x + threadIdx.x;
    if (idx >= NV) return;
    int z = idx % G, y = (idx / G) % G, x = idx / (G * G);
#pragma unroll
    for (int c = 0; c < 3; ++c) {
        const float* f = vel_ws + (size_t)c * NV;
        float lap = grad2(f, x, y, z, 0) + grad2(f, x, y, z, 1) + grad2(f, x, y, z, 2);
        vel_out[(size_t)c * NV + idx] = f[idx] + VISC * DT * lap;
    }
}

// ---------------- host launcher ----------------
extern "C" void kernel_launch(void* const* d_in, const int* in_sizes, int n_in,
                              void* d_out, int out_size, void* d_ws, size_t ws_size,
                              hipStream_t stream) {
    const float* density  = (const float*)d_in[0];
    const float* velocity = (const float*)d_in[1];
    const float* pressure = (const float*)d_in[2];
    const float* sources  = (const float*)d_in[3];
    const float* w1 = (const float*)d_in[4];
    const float* b1 = (const float*)d_in[5];
    const float* w2 = (const float*)d_in[6];
    const float* b2 = (const float*)d_in[7];
    const float* w3 = (const float*)d_in[8];
    const float* b3 = (const float*)d_in[9];
    const float* w4 = (const float*)d_in[10];
    const float* b4 = (const float*)d_in[11];

    float* out = (float*)d_out;                       // [5, G,G,G]
    float* vel_ws = (float*)d_ws;                     // 3*NV floats
    unsigned int* wpack =
        (unsigned int*)((char*)d_ws + (size_t)3 * NV * sizeof(float)); // 58*32*8 dwords

    const int ethreads = 256;
    const int eblocks  = (NV + ethreads - 1) / ethreads;   // 3456

    // K0: pack weights into WMMA B-fragment layout
    pack_weights_kernel<<<58, 256, 0, stream>>>(w1, w2, w3, w4, wpack);

    // K1: advection + sources; density final -> out[0]
    advect_kernel<<<eblocks, ethreads, 0, stream>>>(density, velocity, sources,
                                                    out /* rho out */, vel_ws);

    // K2: turbulence MLP (WMMA), updates vel_ws in place
    mlp_kernel<<<(NV / 16) / 4, 128, 0, stream>>>(vel_ws, out, wpack, b1, b2, b3, b4);

    // K3: pressure update -> out[4]
    pressure_kernel<<<eblocks, ethreads, 0, stream>>>(vel_ws, pressure, out + (size_t)4 * NV);

    // K4: subtract pressure gradient (in place)
    subgrad_kernel<<<eblocks, ethreads, 0, stream>>>(vel_ws, out + (size_t)4 * NV);

    // K5: diffusion -> out[1..3]
    diffuse_kernel<<<eblocks, ethreads, 0, stream>>>(vel_ws, out + (size_t)1 * NV);
}
